// CustomGraphNetJax_51874615001135
// MI455X (gfx1250) — matrix operand order repaired and checked by reading, not compiled, not run on hardware
//
#include <hip/hip_runtime.h>

// ---------------------------------------------------------------------------
// CDNA5 (gfx1250) CustomGraphNet forward pass.
// GEMMs on V_WMMA_F32_16X16X32_BF16 with operand roles swapped:
//   A = pre-swizzled W^T fragments (global, WGP$-resident),
//   B = activation rows (one graph row per lane, contiguous K per lane).
// Kernel is fully specialized per call-site shape so the activation fragment
// build is branch-free straight-line b128 loads + packed bf16 converts.
// ---------------------------------------------------------------------------

typedef __bf16 bf16_t;
typedef __attribute__((ext_vector_type(16))) __bf16 v16bf;
typedef __attribute__((ext_vector_type(8)))  __bf16 v8bf;
typedef __attribute__((ext_vector_type(8)))  float  v8f;

#define NW 8            // waves per block (256 threads)

__device__ __forceinline__ v8f wmma_bf16(v16bf a, v16bf b, v8f c) {
  // (neg_a, A, neg_b, B, c_mod, C, reuse_a, reuse_b)
  return __builtin_amdgcn_wmma_f32_16x16x32_bf16(false, a, false, b, (short)0, c,
                                                 false, false);
}

// Weight fragment (A-matrix layout), pre-swizzled to [kc][nt][lane][16] order:
// one 32-byte aligned contiguous load per lane (2x global_load_b128).
__device__ __forceinline__ v16bf load_w_frag(const bf16_t* wf, int lane, int kc, int nt) {
  return *(const v16bf*)(wf + (size_t)(((kc * 4 + nt) * 32 + lane) << 4));
}

// Fused 2-layer MLP:  out = relu(IN @ W1 + b1) @ W2 + b2
// IN rows = concat( in0[GATHER ? gather[row] : row, 0:K0], in1[row, 0:K1] )
// Computed transposed: D[n, row] = W^T_tile x IN^T_tile, row lives in lanes.
// Requires N % 16 == 0 (always true for this net).
template <int K0, int K1, bool GATHER, bool SCATTER, bool RESID, int DOUT>
__global__ __launch_bounds__(NW * 32) void fused_mlp_kernel(
    const float* __restrict__ in0, const float* __restrict__ in1,
    const int* __restrict__ gather,
    const bf16_t* __restrict__ w1, const float* __restrict__ b1,
    const bf16_t* __restrict__ w2, const float* __restrict__ b2,
    const float* __restrict__ resid, const int* __restrict__ scatter,
    float* __restrict__ out, int N)
{
  static_assert(K1 == 0 || (K0 % 32) == 0, "concat split must be 32-aligned");
  constexpr int K  = K0 + K1;
  constexpr int KP = (K + 31) & ~31;
  constexpr int NK = KP / 32;
  constexpr int OS = SCATTER ? 64 : DOUT;      // output row stride

  __shared__ bf16_t Hs[NW][16 * 64];           // hidden activations, [row][n]
  const int wave = threadIdx.x >> 5;
  const int lane = threadIdx.x & 31;
  const long row0 = ((long)blockIdx.x * NW + wave) * 16;
  if (row0 >= (long)N) return;                 // surplus waves only; tiles are full
  bf16_t* H = Hs[wave];

  __builtin_prefetch(w1, 0, 0);                // global_prefetch of layer-1 frags

  // One graph row per lane (lanes r and r+16 share row r, different K halves).
  const long row = row0 + (lane & 15);
  const long src0 = (GATHER ? (long)gather[row] : row) * (long)K0;
  const long src1 = (K1 > 0) ? row * (long)K1 - K0 : 0;
  const int khalf = (lane & 16) ? 16 : 0;

  // ---- layer 1: D1[n, row] = W1^T x IN^T  (no LDS staging) ----
  v8f acc[4] = {{}, {}, {}, {}};
#pragma unroll
  for (int kc = 0; kc < NK; ++kc) {
    const int kb = kc * 32 + khalf;            // this lane's 16 contiguous k
    float f[16];
#pragma unroll
    for (int g = 0; g < 4; ++g) {
      const int c = kb + g * 4;                // never straddles K0 (32-aligned)
      float4 q = make_float4(0.f, 0.f, 0.f, 0.f);
      if (KP == K || c < K) {                  // folds away unless K < 32
        if (K1 == 0 || kc * 32 < K0)           // folds after unroll
          q = *(const float4*)(in0 + src0 + c);
        else
          q = *(const float4*)(in1 + src1 + c);
      }
      f[g * 4 + 0] = q.x; f[g * 4 + 1] = q.y;
      f[g * 4 + 2] = q.z; f[g * 4 + 3] = q.w;
    }
    v16bf b;
#pragma unroll
    for (int h = 0; h < 16; ++h) b[h] = (bf16_t)f[h];
#pragma unroll
    for (int nt = 0; nt < 4; ++nt)
      acc[nt] = wmma_bf16(load_w_frag(w1, lane, kc, nt), b, acc[nt]);
  }

  // ---- bias + relu; store hidden as [row][n] (16B vector stores) ----
  const int mhi = (lane & 16) ? 8 : 0;         // D: M = r + mhi, N(row) = lane&15
  const int hrow = (lane & 15) * 64;
#pragma unroll
  for (int nt = 0; nt < 4; ++nt) {
    const int n0 = nt * 16 + mhi;
    const float4 bb0 = *(const float4*)(b1 + n0);
    const float4 bb1 = *(const float4*)(b1 + n0 + 4);
    const float bb[8] = {bb0.x, bb0.y, bb0.z, bb0.w, bb1.x, bb1.y, bb1.z, bb1.w};
    v8bf hv;
#pragma unroll
    for (int r = 0; r < 8; ++r) {
      float v = acc[nt][r] + bb[r];
      hv[r] = (bf16_t)(v > 0.f ? v : 0.f);
    }
    *(v8bf*)(H + hrow + n0) = hv;
  }
  asm volatile("s_wait_dscnt 0" ::: "memory"); // cross-lane LDS visibility

  // ---- layer 2: D2[n, row] = W2^T x H^T  (B = 32B contiguous LDS load) ----
  v8f acc2[4] = {{}, {}, {}, {}};
#pragma unroll
  for (int kc = 0; kc < 2; ++kc) {
    v16bf b = *(const v16bf*)(H + hrow + kc * 32 + khalf);
#pragma unroll
    for (int nt = 0; nt < 4; ++nt)
      acc2[nt] = wmma_bf16(load_w_frag(w2, lane, kc, nt), b, acc2[nt]);
  }

  // ---- epilogue: bias, residual, vector store / atomic segment-sum ----
  const long obase = SCATTER ? (long)scatter[row] * OS : row * (long)OS;
#pragma unroll
  for (int nt = 0; nt < 4; ++nt) {
    const int n0 = nt * 16 + mhi;              // 8 consecutive output columns
    if (DOUT == 64 || n0 < DOUT) {             // folds for DOUT==64
      float o[8];
#pragma unroll
      for (int r = 0; r < 8; ++r) o[r] = acc2[nt][r] + b2[n0 + r];
      if (SCATTER) {                           // segment_sum via L2 fp32 atomics
#pragma unroll
        for (int r = 0; r < 8; ++r)
          if (DOUT == 64 || n0 + r < DOUT) atomicAdd(out + obase + n0 + r, o[r]);
      } else if (DOUT == 64) {                 // contiguous: 2x b128
        if (RESID) {
          const float4 r0 = *(const float4*)(resid + obase + n0);
          const float4 r1 = *(const float4*)(resid + obase + n0 + 4);
          o[0] += r0.x; o[1] += r0.y; o[2] += r0.z; o[3] += r0.w;
          o[4] += r1.x; o[5] += r1.y; o[6] += r1.z; o[7] += r1.w;
        }
        *(float4*)(out + obase + n0)     = make_float4(o[0], o[1], o[2], o[3]);
        *(float4*)(out + obase + n0 + 4) = make_float4(o[4], o[5], o[6], o[7]);
      } else {                                 // ragged tail (decoder DOUT=3)
#pragma unroll
        for (int r = 0; r < 8; ++r)
          if (n0 + r < DOUT) out[obase + n0 + r] = o[r];
      }
    }
  }
}

// fp32 W [K, cols] row-major -> bf16 W^T fragments in A-matrix order,
// [kc][nt][lane][16 halves], zero-padded.  A layout used here (contiguous k, same order as B):
// m(=n) = lane&15;  k = kc*32 + (lane>=16 ? 16 : 0) + h.
__global__ void swizzle_w_kernel(const float* __restrict__ src, bf16_t* __restrict__ dst,
                                 int K, int cols, int nkc) {
  int i = blockIdx.x * blockDim.x + threadIdx.x;
  if (i >= nkc * 2048) return;
  const int h    = i & 15;
  const int lane = (i >> 4) & 31;
  const int nt   = (i >> 9) & 3;
  const int kc   = i >> 11;
  const int n = nt * 16 + (lane & 15);
  const int k = kc * 32 + ((lane & 16) ? 16 : 0) + h;
  dst[i] = (bf16_t)((k < K && n < cols) ? src[k * cols + n] : 0.f);
}

__global__ void add_kernel(float* __restrict__ a, const float* __restrict__ b, int n) {
  int i = blockIdx.x * blockDim.x + threadIdx.x;
  if (i < n) a[i] += b[i];
}

// ---------------------------------------------------------------------------
namespace {
struct MlpRaw { const float *W1, *b1, *W2, *b2; };
struct MlpDev { const bf16_t *w1, *w2; const float *b1, *b2; };
}

extern "C" void kernel_launch(void* const* d_in, const int* in_sizes, int n_in,
                              void* d_out, int out_size, void* d_ws, size_t ws_size,
                              hipStream_t stream) {
  (void)in_sizes; (void)n_in; (void)out_size; (void)ws_size;
  enum { N_SP = 20000, N_DN = 50000, E_SP = 160000, E_DN = 400000, E_ML = 150000, L = 64 };

  int idx = 0;
  const float* sparse_x = (const float*)d_in[idx++];
  const float* sparse_e = (const float*)d_in[idx++];
  const float* dense_x  = (const float*)d_in[idx++]; (void)dense_x;  // unused (as in ref)
  const float* dense_e  = (const float*)d_in[idx++];
  const float* multi_e  = (const float*)d_in[idx++];

  auto take = [&]() {
    MlpRaw m;
    m.W1 = (const float*)d_in[idx++]; m.b1 = (const float*)d_in[idx++];
    m.W2 = (const float*)d_in[idx++]; m.b2 = (const float*)d_in[idx++];
    return m;
  };
  // params flattened in JAX pytree order (dict keys sorted alphabetically):
  // decoder, dense_layers[0..7]{message,update}, enc_dense_e, enc_dense_x,
  // enc_multi_e, enc_sparse_e, enc_sparse_x, link{message,update},
  // sparse_layers[0..7]{message,update}
  MlpRaw r_dec = take();
  MlpRaw r_dm[8], r_du[8];
  for (int p = 0; p < 8; ++p) { r_dm[p] = take(); r_du[p] = take(); }
  MlpRaw r_ede = take();
  MlpRaw r_edx = take(); (void)r_edx;
  MlpRaw r_eme = take();
  MlpRaw r_ese = take();
  MlpRaw r_esx = take();
  MlpRaw r_lm = take(), r_lu = take();
  MlpRaw r_sm[8], r_su[8];
  for (int p = 0; p < 8; ++p) { r_sm[p] = take(); r_su[p] = take(); }
  const int* sp_ei = (const int*)d_in[idx++];
  const int* dn_ei = (const int*)d_in[idx++];
  const int* ml_ei = (const int*)d_in[idx++];

  // ---- workspace carve-out ----
  float* ws = (float*)d_ws;
  size_t off = 0;
  auto fal = [&](size_t n) { float* p = ws + off; off += n; return p; };
  float* nls  = fal((size_t)N_SP * L);
  float* u    = fal((size_t)N_SP * L);
  float* els  = fal((size_t)E_SP * L);
  float* elm  = fal((size_t)E_ML * L);
  float* eld  = fal((size_t)E_DN * L);   // 102 MB, mostly L2-resident across passes
  float* aggS = fal((size_t)N_SP * L);
  float* me   = fal((size_t)E_ML * L);   // also the [N_DN, 192] reshaped view
  float* d0   = fal((size_t)N_DN * L);
  float* v    = fal((size_t)N_DN * L);
  float* aggD = fal((size_t)N_DN * L);
  off = (off + 7) & ~(size_t)7;          // 32B-align the bf16 fragment pool
  bf16_t* wpool = (bf16_t*)(ws + off);
  size_t woff = 0;
  auto bal = [&](size_t n) { bf16_t* p = wpool + woff; woff += n; return p; };

  auto prep = [&](const MlpRaw& m, int K, int dout) {
    MlpDev d;
    const int nkc1 = (K + 31) >> 5;
    bf16_t* w1 = bal((size_t)nkc1 * 2048);   // frag-order, 32B aligned
    bf16_t* w2 = bal((size_t)2 * 2048);
    hipLaunchKernelGGL(swizzle_w_kernel, dim3((nkc1 * 2048 + 255) / 256), dim3(256),
                       0, stream, m.W1, w1, K, 64, nkc1);
    hipLaunchKernelGGL(swizzle_w_kernel, dim3((2 * 2048 + 255) / 256), dim3(256),
                       0, stream, m.W2, w2, 64, dout, 2);
    d.w1 = w1; d.w2 = w2; d.b1 = m.b1; d.b2 = m.b2;
    return d;
  };

  MlpDev M_esx = prep(r_esx, 16, 64);
  MlpDev M_ese = prep(r_ese, 8, 64);
  MlpDev M_eme = prep(r_eme, 8, 64);
  MlpDev M_ede = prep(r_ede, 8, 64);
  MlpDev M_sm[8], M_su[8], M_dm[8], M_du[8];
  for (int p = 0; p < 8; ++p) {
    M_sm[p] = prep(r_sm[p], 128, 64); M_su[p] = prep(r_su[p], 128, 64);
    M_dm[p] = prep(r_dm[p], 128, 64); M_du[p] = prep(r_du[p], 128, 64);
  }
  MlpDev M_lm = prep(r_lm, 128, 64), M_lu = prep(r_lu, 192, 64);
  MlpDev M_dec = prep(r_dec, 64, 3);

  auto grid = [](int N) { return dim3((unsigned)((N / 16 + NW - 1) / NW)); };
  const dim3 blk(NW * 32);

  // ---- encoders ----
  fused_mlp_kernel<16, 0, false, false, false, 64><<<grid(N_SP), blk, 0, stream>>>(
      sparse_x, nullptr, nullptr, M_esx.w1, M_esx.b1, M_esx.w2, M_esx.b2,
      nullptr, nullptr, nls, N_SP);
  fused_mlp_kernel<8, 0, false, false, false, 64><<<grid(E_SP), blk, 0, stream>>>(
      sparse_e, nullptr, nullptr, M_ese.w1, M_ese.b1, M_ese.w2, M_ese.b2,
      nullptr, nullptr, els, E_SP);
  fused_mlp_kernel<8, 0, false, false, false, 64><<<grid(E_ML), blk, 0, stream>>>(
      multi_e, nullptr, nullptr, M_eme.w1, M_eme.b1, M_eme.w2, M_eme.b2,
      nullptr, nullptr, elm, E_ML);
  fused_mlp_kernel<8, 0, false, false, false, 64><<<grid(E_DN), blk, 0, stream>>>(
      dense_e, nullptr, nullptr, M_ede.w1, M_ede.b1, M_ede.w2, M_ede.b2,
      nullptr, nullptr, eld, E_DN);

  // ---- sparse message passing (8 passes, residual) ----
  hipMemcpyAsync(u, nls, (size_t)N_SP * L * 4, hipMemcpyDeviceToDevice, stream);
  for (int p = 0; p < 8; ++p) {
    hipMemsetAsync(aggS, 0, (size_t)N_SP * L * 4, stream);
    fused_mlp_kernel<64, 64, true, true, false, 64><<<grid(E_SP), blk, 0, stream>>>(
        u, els, sp_ei, M_sm[p].w1, M_sm[p].b1, M_sm[p].w2, M_sm[p].b2,
        nullptr, sp_ei + E_SP, aggS, E_SP);                     // msg + segment_sum
    fused_mlp_kernel<64, 64, false, false, true, 64><<<grid(N_SP), blk, 0, stream>>>(
        u, aggS, nullptr, M_su[p].w1, M_su[p].b1, M_su[p].w2, M_su[p].b2,
        u, nullptr, u, N_SP);                                   // residual update
  }
  { int n = N_SP * L;
    hipLaunchKernelGGL(add_kernel, dim3((n + 255) / 256), dim3(256), 0, stream, u, nls, n); }

  // ---- link layer: message then fixed-fanin (reshape) aggregate ----
  fused_mlp_kernel<64, 64, true, false, false, 64><<<grid(E_ML), blk, 0, stream>>>(
      u, elm, ml_ei, M_lm.w1, M_lm.b1, M_lm.w2, M_lm.b2,
      nullptr, nullptr, me, E_ML);
  fused_mlp_kernel<192, 0, false, false, false, 64><<<grid(N_DN), blk, 0, stream>>>(
      me, nullptr, nullptr, M_lu.w1, M_lu.b1, M_lu.w2, M_lu.b2,
      nullptr, nullptr, d0, N_DN);

  // ---- dense message passing (8 passes, residual) ----
  hipMemcpyAsync(v, d0, (size_t)N_DN * L * 4, hipMemcpyDeviceToDevice, stream);
  for (int p = 0; p < 8; ++p) {
    hipMemsetAsync(aggD, 0, (size_t)N_DN * L * 4, stream);
    fused_mlp_kernel<64, 64, true, true, false, 64><<<grid(E_DN), blk, 0, stream>>>(
        v, eld, dn_ei, M_dm[p].w1, M_dm[p].b1, M_dm[p].w2, M_dm[p].b2,
        nullptr, dn_ei + E_DN, aggD, E_DN);
    fused_mlp_kernel<64, 64, false, false, true, 64><<<grid(N_DN), blk, 0, stream>>>(
        v, aggD, nullptr, M_du[p].w1, M_du[p].b1, M_du[p].w2, M_du[p].b2,
        v, nullptr, v, N_DN);
  }
  { int n = N_DN * L;
    hipLaunchKernelGGL(add_kernel, dim3((n + 255) / 256), dim3(256), 0, stream, v, d0, n); }

  // ---- decoder ----
  fused_mlp_kernel<64, 0, false, false, false, 3><<<grid(N_DN), blk, 0, stream>>>(
      v, nullptr, nullptr, M_dec.w1, M_dec.b1, M_dec.w2, M_dec.b2,
      nullptr, nullptr, (float*)d_out, N_DN);
}
